// SNN_Model_14224931684742
// MI455X (gfx1250) — matrix operand order repaired and checked by reading, not compiled, not run on hardware
//
#include <hip/hip_runtime.h>
#include <hip/hip_bf16.h>
#include <cmath>

#define THRESH 0.3f
#define MEMD   0.6f
#define BATCH  128

typedef __attribute__((ext_vector_type(16))) _Float16 v16h;
typedef __attribute__((ext_vector_type(8)))  float    v8f;
typedef __attribute__((ext_vector_type(8)))  _Float16 v8h;

union V16H { v16h v; v8h h[2]; };

// ============================================================================
// WMMA GEMM:  D[M][N] (f32 row-major) = A[M][K] (f16 row-major) x Bt[N][K]^T
// Each wave computes MB vertically-stacked 16x16 tiles (MB*16 x 16 output),
// sharing one B fragment across MB v_wmma_f32_16x16x32_f16 ops per K-step of
// 32. Statically-indexed ping-pong register buffers (slot0/slot1, unroll-by-2
// K loop) keep all fragments in VGPRs (no scratch/LDS spill) while loads for
// K-step k+64 are in flight during the WMMAs of K-step k.
// Requires M%(16*MB)==0, N%16==0, K%64==0, K>=128 (all call sites satisfy).
// Fragment layouts per CDNA5 ISA 7.12.2:
//   A 16x32 f16 : lane l holds row M=l%16; halves 0-7 -> K = (l>=16?8:0)+0..7,
//                 halves 8-15 -> K = 16+(l>=16?8:0)+0..7          (2x b128 loads)
//   B 32x16 f16 : lane l holds col N=l%16; same K mapping as A (2x b128 loads)
//   D 16x16 f32 : vgpr r: lanes0-15 (M=r,N=lane), lanes16-31 (M=8+r,N=lane-16)
// ============================================================================
template <int MB>
__global__ __launch_bounds__(256) void k_gemm_wmma(
    const _Float16* __restrict__ A, const _Float16* __restrict__ Bt,
    float* __restrict__ D, int M, int N, int K)
{
  const int lane    = threadIdx.x & 31;
  const int wave    = threadIdx.x >> 5;
  const int mGroups = M / (16 * MB);
  const int nTiles  = N >> 4;
  const int g       = blockIdx.x * 8 + wave;
  if (g >= mGroups * nTiles) return;          // wave-uniform: EXEC stays full
  const int m0 = (g % mGroups) * (16 * MB);
  const int n0 = (g / mGroups) << 4;

  const _Float16* aBase = A  + (size_t)(m0 + (lane & 15)) * K + ((lane >> 4) << 3);
  const _Float16* bBase = Bt + (size_t)(n0 + (lane & 15)) * K + ((lane >> 4) << 3);

  v8f  acc[MB] = {};
  V16H a0[MB], a1[MB];
  V16H b0, b1;

  auto loadSlot = [&](V16H (&aS)[MB], V16H& bS, int k0) {
    bS.h[0] = *(const v8h*)(bBase + k0);
    bS.h[1] = *(const v8h*)(bBase + k0 + 16);
#pragma unroll
    for (int t = 0; t < MB; ++t) {
      const _Float16* ap = aBase + (size_t)t * 16 * K + k0;
      aS[t].h[0] = *(const v8h*)(ap);
      aS[t].h[1] = *(const v8h*)(ap + 16);
    }
  };
  auto mmaSlot = [&](V16H (&aS)[MB], V16H& bS) {
#pragma unroll
    for (int t = 0; t < MB; ++t) {
      acc[t] = __builtin_amdgcn_wmma_f32_16x16x32_f16(false, aS[t].v, false, bS.v,
                                                      (short)0, acc[t], false, false);
    }
  };

  loadSlot(a0, b0, 0);
  loadSlot(a1, b1, 32);
  int k0 = 0;
  for (; k0 + 128 <= K; k0 += 64) {
    mmaSlot(a0, b0);
    loadSlot(a0, b0, k0 + 64);                 // refill slot0 while slot1 computes
    __builtin_prefetch(bBase + k0 + 128, 0, 1);
    mmaSlot(a1, b1);
    loadSlot(a1, b1, k0 + 96);
    __builtin_prefetch(aBase + k0 + 128, 0, 1);
  }
  mmaSlot(a0, b0);                             // epilogue: last two K-steps
  mmaSlot(a1, b1);

  const int mLane = ((lane >> 4) << 3);
  const int nLane = (lane & 15);
#pragma unroll
  for (int t = 0; t < MB; ++t) {
    float* dPtr = D + (size_t)(m0 + t * 16 + mLane) * N + (n0 + nLane);
#pragma unroll
    for (int r = 0; r < 8; ++r)
      __builtin_nontemporal_store(acc[t][r], dPtr + (size_t)r * N);
  }
}

// ---------------------------------------------------------------------------
// conv1 (1->128, 3x3 pad 1) fused with its LIF update. x is constant per step.
// idx over B*128*28*28
// ---------------------------------------------------------------------------
__global__ void k_conv1_lif(const float* __restrict__ x, const float* __restrict__ w1,
                            const float* __restrict__ b1, float* __restrict__ mem,
                            float* __restrict__ spk, int total)
{
  int idx = blockIdx.x * blockDim.x + threadIdx.x;
  if (idx >= total) return;
  int p = idx % 784;
  int c = (idx / 784) & 127;
  int b = idx / (784 * 128);
  int h = p / 28, w = p % 28;
  const float* xb = x + (size_t)b * 784;
  const float* wc = w1 + c * 9;
  float conv = b1[c];
#pragma unroll
  for (int kh = 0; kh < 3; ++kh) {
    int ih = h + kh - 1;
    if (ih < 0 || ih >= 28) continue;
#pragma unroll
    for (int kw = 0; kw < 3; ++kw) {
      int iw = w + kw - 1;
      if (iw < 0 || iw >= 28) continue;
      conv += xb[ih * 28 + iw] * wc[kh * 3 + kw];
    }
  }
  float m = (mem[idx] - THRESH * spk[idx]) * MEMD + conv;
  mem[idx] = m;
  spk[idx] = (m > THRESH) ? 1.0f : 0.0f;
}

// ---------------------------------------------------------------------------
// im2col with fused 2x2 avg-pool of spikes and decay scale.
// Produces col[N][K] (patch contiguous) f16.  N = B*HO*WO, K = C*9.
// spk is [B][C][Hs][Ws] with Hs=2*HO, Ws=2*WO.
// ---------------------------------------------------------------------------
__global__ void k_im2col_pool(const float* __restrict__ spk, _Float16* __restrict__ col,
                              int C, int HO, int WO, int Ws, int HWs, int Npix,
                              float scale, int total)
{
  int e = blockIdx.x * blockDim.x + threadIdx.x;
  if (e >= total) return;
  const int K = C * 9;
  int n = e / K;
  int k = e - n * K;
  int b  = n / Npix;
  int p  = n - b * Npix;
  int oh = p / WO, ow = p - oh * WO;
  int ic = k / 9;
  int r  = k - ic * 9;
  int kh = r / 3, kw = r - kh * 3;
  int ih = oh + kh - 1, iw = ow + kw - 1;
  float v = 0.0f;
  if (ih >= 0 && ih < HO && iw >= 0 && iw < WO) {
    const float* sp = spk + ((size_t)b * C + ic) * HWs + (2 * ih) * Ws + 2 * iw;
    v = (sp[0] + sp[1] + sp[Ws] + sp[Ws + 1]) * scale;
  }
  col[e] = (_Float16)v;
}

// LIF for conv2/conv3: convout is [C][B*HW] from the GEMM, states are NCHW.
__global__ void k_lif_conv(float* __restrict__ mem, float* __restrict__ spk,
                           const float* __restrict__ convout, const float* __restrict__ bias,
                           int C, int HW, int total)
{
  int idx = blockIdx.x * blockDim.x + threadIdx.x;
  if (idx >= total) return;
  int p = idx % HW;
  int c = (idx / HW) % C;
  int b = idx / (HW * C);
  float pre = convout[(size_t)c * (BATCH * HW) + (size_t)b * HW + p] + bias[c];
  float m = (mem[idx] - THRESH * spk[idx]) * MEMD + pre;
  mem[idx] = m;
  spk[idx] = (m > THRESH) ? 1.0f : 0.0f;
}

// pool3 + flatten + decay: x_fc16[b][k] and xbT16[k][b] = x_fc*beta1 transposed.
__global__ void k_pool3(const float* __restrict__ spk3, _Float16* __restrict__ x_fc16,
                        _Float16* __restrict__ xbT16, const float* __restrict__ beta1,
                        float decay, int total)
{
  int idx = blockIdx.x * blockDim.x + threadIdx.x;
  if (idx >= total) return;
  int b = idx / 2304;
  int k = idx - b * 2304;
  int c = k / 9;
  int r = k - c * 9;
  int ph = r / 3, pw = r - ph * 3;
  const float* sp = spk3 + ((size_t)b * 256 + c) * 49 + (2 * ph) * 7 + 2 * pw;
  float v = (sp[0] + sp[1] + sp[7] + sp[8]) * 0.25f * decay;
  x_fc16[(size_t)b * 2304 + k] = (_Float16)v;
  xbT16[(size_t)k * 128 + b]  = (_Float16)(v * beta1[k]);
}

// fc1 LIF: state = Wx + b + alpha*(x@hebb). Also emits f16 operands for the
// Hebbian outer-product GEMM (yT16[j][b]) and fc2 GEMM (hspk16[b][j]).
__global__ void k_lif_fc1(float* __restrict__ h1_mem, float* __restrict__ h1_spk,
                          const float* __restrict__ s_w, const float* __restrict__ s_h,
                          const float* __restrict__ fc1b, const float* __restrict__ alpha1,
                          const float* __restrict__ eta1, _Float16* __restrict__ yT16,
                          _Float16* __restrict__ hspk16, float decay, int total)
{
  int idx = blockIdx.x * blockDim.x + threadIdx.x;
  if (idx >= total) return;
  int b = idx / 512;
  int j = idx - b * 512;
  float st = s_w[(size_t)j * 128 + b] + fc1b[j] + alpha1[0] * s_h[(size_t)j * 128 + b];
  float m = (h1_mem[idx] - THRESH * h1_spk[idx]) * MEMD + st;
  h1_mem[idx] = m;
  float s = (m > THRESH) ? 1.0f : 0.0f;
  h1_spk[idx] = s;
  yT16[(size_t)j * 128 + b] = (_Float16)(m * (1.0f / THRESH) - eta1[j]);
  hspk16[idx] = (_Float16)(s * decay);
}

// hebb1 = clip(0.95*hebb1 - outer/B, -2, 2); refresh f16 transposed mirror.
__global__ void k_hebb_update(float* __restrict__ hebb, const float* __restrict__ outer,
                              _Float16* __restrict__ hebbT16, int total)
{
  int idx = blockIdx.x * blockDim.x + threadIdx.x;
  if (idx >= total) return;
  int k = idx / 512;
  int j = idx - k * 512;
  float h = 0.95f * hebb[idx] - outer[idx] * (1.0f / 128.0f);
  h = fminf(2.0f, fmaxf(-2.0f, h));
  hebb[idx] = h;
  hebbT16[(size_t)j * 2304 + k] = (_Float16)h;
}

__global__ void k_lif_h2(float* __restrict__ mem, float* __restrict__ spk,
                         float* __restrict__ ssum, const float* __restrict__ h2pre,
                         const float* __restrict__ fc2b, int total)
{
  int idx = blockIdx.x * blockDim.x + threadIdx.x;
  if (idx >= total) return;
  int b = idx / 10;
  int o = idx - b * 10;
  float pre = h2pre[(size_t)o * 128 + b] + fc2b[o];
  float m = (mem[idx] - THRESH * spk[idx]) * MEMD + pre;
  mem[idx] = m;
  float s = (m > THRESH) ? 1.0f : 0.0f;
  spk[idx] = s;
  ssum[idx] += s;
}

__global__ void k_f32_to_f16(const float* __restrict__ in, _Float16* __restrict__ out, int n)
{
  int i = blockIdx.x * blockDim.x + threadIdx.x;
  if (i < n) out[i] = (_Float16)in[i];
}

// fc2 weights (10x512) zero-padded to 16x512 so the WMMA GEMM needs no guards.
__global__ void k_fc2w_pad(const float* __restrict__ w, _Float16* __restrict__ out, int n)
{
  int i = blockIdx.x * blockDim.x + threadIdx.x;
  if (i >= n) return;
  int r = i / 512;
  out[i] = (r < 10) ? (_Float16)w[i] : (_Float16)0.0f;
}

__global__ void k_hebb_init(const float* __restrict__ hin, float* __restrict__ hebb,
                            _Float16* __restrict__ hebbT16, int total)
{
  int idx = blockIdx.x * blockDim.x + threadIdx.x;
  if (idx >= total) return;
  float v = hin[idx];
  hebb[idx] = v;
  int k = idx / 512;
  int j = idx - k * 512;
  hebbT16[(size_t)j * 2304 + k] = (_Float16)v;
}

__global__ void k_copy_scale(float* __restrict__ dst, const float* __restrict__ src,
                             int n, float scale)
{
  int i = blockIdx.x * blockDim.x + threadIdx.x;
  if (i < n) dst[i] = src[i] * scale;
}

__global__ void k_mean(const float* __restrict__ src, float* __restrict__ dst, int n)
{
  if (blockIdx.x == 0 && threadIdx.x == 0) {
    float s = 0.0f;
    for (int i = 0; i < n; ++i) s += src[i];
    *dst = s / (float)n;
  }
}

// ============================================================================
extern "C" void kernel_launch(void* const* d_in, const int* in_sizes, int n_in,
                              void* d_out, int out_size, void* d_ws, size_t ws_size,
                              hipStream_t stream) {
  (void)in_sizes; (void)n_in; (void)out_size; (void)ws_size;

  const float* x        = (const float*)d_in[0];
  const float* hebb1_in = (const float*)d_in[1];
  const float* hebb2_in = (const float*)d_in[2];
  const float* w1   = (const float*)d_in[3];
  const float* b1   = (const float*)d_in[4];
  const float* w2   = (const float*)d_in[5];
  const float* b2   = (const float*)d_in[6];
  const float* w3   = (const float*)d_in[7];
  const float* b3   = (const float*)d_in[8];
  const float* fc1w = (const float*)d_in[9];
  const float* fc1b = (const float*)d_in[10];
  const float* fc2w = (const float*)d_in[11];
  const float* fc2b = (const float*)d_in[12];
  const float* alpha1 = (const float*)d_in[13];
  const float* beta1  = (const float*)d_in[14];
  const float* eta1   = (const float*)d_in[15];
  float* out = (float*)d_out;

  // ---- workspace bump allocator (256B aligned) ----
  char* p = (char*)d_ws;
  auto alloc = [&](size_t bytes) -> char* {
    char* r = p;
    p += (bytes + 255) & ~(size_t)255;
    return r;
  };
  const int C1 = 12845056;  // 128*128*28*28
  const int C2 = 6422528;   // 128*256*14*14
  const int C3 = 1605632;   // 128*256*7*7
  float*    c1_mem  = (float*)   alloc((size_t)C1 * 4);
  float*    c1_spk  = (float*)   alloc((size_t)C1 * 4);
  float*    c2_mem  = (float*)   alloc((size_t)C2 * 4);
  float*    c2_spk  = (float*)   alloc((size_t)C2 * 4);
  float*    c3_mem  = (float*)   alloc((size_t)C3 * 4);
  float*    c3_spk  = (float*)   alloc((size_t)C3 * 4);
  _Float16* col     = (_Float16*)alloc((size_t)28901376 * 2); // max im2col (conv2)
  float*    convout = (float*)   alloc((size_t)C2 * 4);       // max 256*25088
  _Float16* w2_16   = (_Float16*)alloc((size_t)294912 * 2);
  _Float16* w3_16   = (_Float16*)alloc((size_t)589824 * 2);
  _Float16* fc1w16  = (_Float16*)alloc((size_t)1179648 * 2);
  _Float16* fc2w16  = (_Float16*)alloc((size_t)8192 * 2);
  float*    hebb    = (float*)   alloc((size_t)1179648 * 4);
  _Float16* hebbT16 = (_Float16*)alloc((size_t)1179648 * 2);
  float*    outer   = (float*)   alloc((size_t)1179648 * 4);
  _Float16* x_fc16  = (_Float16*)alloc((size_t)294912 * 2);
  _Float16* xbT16   = (_Float16*)alloc((size_t)294912 * 2);
  float*    s_w     = (float*)   alloc((size_t)65536 * 4);
  float*    s_h     = (float*)   alloc((size_t)65536 * 4);
  float*    h1_mem  = (float*)   alloc((size_t)65536 * 4);
  float*    h1_spk  = (float*)   alloc((size_t)65536 * 4);
  _Float16* yT16    = (_Float16*)alloc((size_t)65536 * 2);
  _Float16* hspk16  = (_Float16*)alloc((size_t)65536 * 2);
  float*    h2pre   = (float*)   alloc((size_t)2048 * 4);
  float*    h2_mem  = (float*)   alloc((size_t)1280 * 4);
  float*    h2_spk  = (float*)   alloc((size_t)1280 * 4);
  float*    h2_sum  = (float*)   alloc((size_t)1280 * 4);

#define GRID(n) dim3(((n) + 255) / 256)
  // GEMM grid: tiles-groups / 8 waves per block
#define GGRID(mGroups, nTiles) dim3((((mGroups) * (nTiles)) + 7) / 8)

  // ---- per-call state init (deterministic) ----
  hipMemsetAsync(c1_mem, 0, (size_t)C1 * 4, stream);
  hipMemsetAsync(c1_spk, 0, (size_t)C1 * 4, stream);
  hipMemsetAsync(c2_mem, 0, (size_t)C2 * 4, stream);
  hipMemsetAsync(c2_spk, 0, (size_t)C2 * 4, stream);
  hipMemsetAsync(c3_mem, 0, (size_t)C3 * 4, stream);
  hipMemsetAsync(c3_spk, 0, (size_t)C3 * 4, stream);
  hipMemsetAsync(h1_mem, 0, (size_t)65536 * 4, stream);
  hipMemsetAsync(h1_spk, 0, (size_t)65536 * 4, stream);
  hipMemsetAsync(h2_mem, 0, (size_t)1280 * 4, stream);
  hipMemsetAsync(h2_spk, 0, (size_t)1280 * 4, stream);
  hipMemsetAsync(h2_sum, 0, (size_t)1280 * 4, stream);

  // ---- weight precision conversion (once) ----
  k_f32_to_f16<<<GRID(294912), 256, 0, stream>>>(w2, w2_16, 294912);
  k_f32_to_f16<<<GRID(589824), 256, 0, stream>>>(w3, w3_16, 589824);
  k_f32_to_f16<<<GRID(1179648), 256, 0, stream>>>(fc1w, fc1w16, 1179648);
  k_fc2w_pad<<<GRID(8192), 256, 0, stream>>>(fc2w, fc2w16, 8192);
  k_hebb_init<<<GRID(1179648), 256, 0, stream>>>(hebb1_in, hebb, hebbT16, 1179648);

  // ---- T timesteps ----
  for (int step = 0; step < 10; ++step) {
    float decay = expf(-(float)step / 40.0f);

    // layer 1: conv1 + LIF (direct f32, tiny)
    k_conv1_lif<<<GRID(C1), 256, 0, stream>>>(x, w1, b1, c1_mem, c1_spk, C1);

    // layer 2: pool(c1_spk)*decay -> im2col -> WMMA GEMM (64-row blocks) -> LIF
    k_im2col_pool<<<GRID(28901376), 256, 0, stream>>>(c1_spk, col, 128, 14, 14, 28,
                                                      784, 196, 0.25f * decay, 28901376);
    k_gemm_wmma<4><<<GGRID(4, 1568), 256, 0, stream>>>(w2_16, col, convout,
                                                       256, 25088, 1152);
    k_lif_conv<<<GRID(C2), 256, 0, stream>>>(c2_mem, c2_spk, convout, b2, 256, 196, C2);

    // layer 3
    k_im2col_pool<<<GRID(14450688), 256, 0, stream>>>(c2_spk, col, 256, 7, 7, 14,
                                                      196, 49, 0.25f * decay, 14450688);
    k_gemm_wmma<4><<<GGRID(4, 392), 256, 0, stream>>>(w3_16, col, convout,
                                                      256, 6272, 2304);
    k_lif_conv<<<GRID(C3), 256, 0, stream>>>(c3_mem, c3_spk, convout, b3, 256, 49, C3);

    // flatten: x_fc (f16) and (x_fc*beta1)^T (f16)
    k_pool3<<<GRID(294912), 256, 0, stream>>>(c3_spk, x_fc16, xbT16, beta1, decay, 294912);

    // fc1: W*x and x@hebb (uses hebbT16 from previous step) -> LIF
    k_gemm_wmma<4><<<GGRID(8, 8), 256, 0, stream>>>(fc1w16, x_fc16, s_w, 512, 128, 2304);
    k_gemm_wmma<4><<<GGRID(8, 8), 256, 0, stream>>>(hebbT16, x_fc16, s_h, 512, 128, 2304);
    k_lif_fc1<<<GRID(65536), 256, 0, stream>>>(h1_mem, h1_spk, s_w, s_h, fc1b, alpha1,
                                               eta1, yT16, hspk16, decay, 65536);

    // Hebbian fast-weight update: outer = (x*beta1)^T @ (mem/TH - eta1), then decay+clip
    k_gemm_wmma<4><<<GGRID(36, 32), 256, 0, stream>>>(xbT16, yT16, outer, 2304, 512, 128);
    k_hebb_update<<<GRID(1179648), 256, 0, stream>>>(hebb, outer, hebbT16, 1179648);

    // fc2 (M padded to 16, single m-tile -> MB=1) -> LIF + spike accumulation
    k_gemm_wmma<1><<<GGRID(1, 8), 256, 0, stream>>>(fc2w16, hspk16, h2pre, 16, 128, 512);
    k_lif_h2<<<GRID(1280), 256, 0, stream>>>(h2_mem, h2_spk, h2_sum, h2pre, fc2b, 1280);
  }

  // ---- outputs: [outs(1280) | mean(1) | hebb1(1179648) | hebb2(5120)] ----
  k_copy_scale<<<GRID(1280), 256, 0, stream>>>(out, h2_mem, 1280, 1.0f / THRESH);
  k_mean<<<dim3(1), 32, 0, stream>>>(h2_sum, out + 1280, 1280);
  hipMemcpyAsync(out + 1281, hebb, (size_t)1179648 * 4, hipMemcpyDeviceToDevice, stream);
  hipMemcpyAsync(out + 1281 + 1179648, hebb2_in, (size_t)5120 * 4,
                 hipMemcpyDeviceToDevice, stream);
#undef GRID
#undef GGRID
}
